// DRHPAttention_83116207112512
// MI455X (gfx1250) — matrix operand order repaired and checked by reading, not compiled, not run on hardware
//
#include <hip/hip_runtime.h>
#include <hip/hip_bf16.h>
#include <math.h>

// ---------------------------------------------------------------------------
// Types / helpers
// ---------------------------------------------------------------------------
typedef __attribute__((ext_vector_type(16))) __bf16 v16bf;
typedef __attribute__((ext_vector_type(8)))  float  v8f;
typedef __attribute__((ext_vector_type(4)))  unsigned int v4u;
typedef unsigned short u16;
typedef unsigned int   u32;

#define NPTS    8000      // D*H*W = 20^3
#define CCH     16
#define DMQ     32
#define ODM     16
#define TOPK    2400      // int(0.3 * 8000)
#define NTILES  500       // 8000 / 16
#define SSTRIDE 8016      // padded LDS score row stride (bf16 elems)
#define INV_SQRT_DM 0.17677669529663687f

__device__ __forceinline__ u16 f2bf(float f) {
  u32 u = __float_as_uint(f);
  u32 r = (u + 0x7FFFu + ((u >> 16) & 1u)) >> 16;   // round-to-nearest-even
  return (u16)r;
}
__device__ __forceinline__ float bf2f(u16 u) { return __uint_as_float(((u32)u) << 16); }

union Frag { v16bf bf; v4u q[2]; u16 us[16]; };

__device__ __forceinline__ v8f wmma_bf16(v16bf a, v16bf b, v8f c) {
  // D(16x16,f32) = A(16x32,bf16) x B(32x16,bf16) + C
  return __builtin_amdgcn_wmma_f32_16x16x32_bf16(false, a, false, b, (short)0, c, false, false);
}

// ---------------------------------------------------------------------------
// Phase A: conv stride-1  (s1[o][n], 16 x 8000)
// ---------------------------------------------------------------------------
__global__ void conv1_k(const float* __restrict__ x, const float* __restrict__ w,
                        const float* __restrict__ b, float* __restrict__ s1) {
  int idx = blockIdx.x * blockDim.x + threadIdx.x;
  if (idx >= CCH * NPTS) return;
  int o = idx / NPTS, n = idx % NPTS;
  int z = n / 400, y = (n / 20) % 20, x3 = n % 20;
  float acc = b[o];
  for (int i = 0; i < CCH; ++i) {
    const float* xi = x + i * NPTS;
    const float* wi = w + (o * CCH + i) * 27;
    for (int dz = -1; dz <= 1; ++dz) {
      int zz = z + dz; if (zz < 0 || zz > 19) continue;
      for (int dy = -1; dy <= 1; ++dy) {
        int yy = y + dy; if (yy < 0 || yy > 19) continue;
        for (int dx = -1; dx <= 1; ++dx) {
          int xx = x3 + dx; if (xx < 0 || xx > 19) continue;
          acc += xi[zz * 400 + yy * 20 + xx] * wi[(dz + 1) * 9 + (dy + 1) * 3 + (dx + 1)];
        }
      }
    }
  }
  s1[idx] = acc;
}

// ---------------------------------------------------------------------------
// Phase A: conv stride-2  (s2[o][zo*100+yo*10+xo], 16 x 1000)
// ---------------------------------------------------------------------------
__global__ void conv2_k(const float* __restrict__ x, const float* __restrict__ w,
                        const float* __restrict__ b, float* __restrict__ s2) {
  int idx = blockIdx.x * blockDim.x + threadIdx.x;
  if (idx >= CCH * 1000) return;
  int o = idx / 1000, n = idx % 1000;
  int zo = n / 100, yo = (n / 10) % 10, xo = n % 10;
  float acc = b[o];
  for (int i = 0; i < CCH; ++i) {
    const float* xi = x + i * NPTS;
    const float* wi = w + (o * CCH + i) * 27;
    for (int dz = 0; dz < 3; ++dz) {
      int zz = 2 * zo - 1 + dz; if (zz < 0 || zz > 19) continue;
      for (int dy = 0; dy < 3; ++dy) {
        int yy = 2 * yo - 1 + dy; if (yy < 0 || yy > 19) continue;
        for (int dx = 0; dx < 3; ++dx) {
          int xx = 2 * xo - 1 + dx; if (xx < 0 || xx > 19) continue;
          acc += xi[zz * 400 + yy * 20 + xx] * wi[dz * 9 + dy * 3 + dx];
        }
      }
    }
  }
  s2[idx] = acc;
}

// ---------------------------------------------------------------------------
// Phase A: mf = 0.5*s1 + 0.5*trilinear_up(s2), prune |.|<1e-3   (mf[c][n])
// ---------------------------------------------------------------------------
__global__ void combine_k(const float* __restrict__ s1, const float* __restrict__ s2,
                          float* __restrict__ mf) {
  int idx = blockIdx.x * blockDim.x + threadIdx.x;
  if (idx >= CCH * NPTS) return;
  int c = idx / NPTS, n = idx % NPTS;
  int z = n / 400, y = (n / 20) % 20, x = n % 20;
  // half-pixel trilinear from 10^3: src = (d + 0.5)*0.5 - 0.5
  float fz = 0.5f * z - 0.25f, fy = 0.5f * y - 0.25f, fx = 0.5f * x - 0.25f;
  int iz = (int)floorf(fz), iy = (int)floorf(fy), ix = (int)floorf(fx);
  float tz = fz - iz, ty = fy - iy, tx = fx - ix;
  int z0 = max(iz, 0), z1 = min(iz + 1, 9);
  int y0 = max(iy, 0), y1 = min(iy + 1, 9);
  int x0 = max(ix, 0), x1 = min(ix + 1, 9);
  const float* sc = s2 + c * 1000;
  float v000 = sc[z0 * 100 + y0 * 10 + x0], v001 = sc[z0 * 100 + y0 * 10 + x1];
  float v010 = sc[z0 * 100 + y1 * 10 + x0], v011 = sc[z0 * 100 + y1 * 10 + x1];
  float v100 = sc[z1 * 100 + y0 * 10 + x0], v101 = sc[z1 * 100 + y0 * 10 + x1];
  float v110 = sc[z1 * 100 + y1 * 10 + x0], v111 = sc[z1 * 100 + y1 * 10 + x1];
  float a00 = v000 + tx * (v001 - v000), a01 = v010 + tx * (v011 - v010);
  float a10 = v100 + tx * (v101 - v100), a11 = v110 + tx * (v111 - v110);
  float b0 = a00 + ty * (a01 - a00), b1 = a10 + ty * (a11 - a10);
  float up = b0 + tz * (b1 - b0);
  float m = 0.5f * s1[idx] + 0.5f * up;
  if (fabsf(m) < 0.001f) m = 0.0f;
  mf[idx] = m;
}

// ---------------------------------------------------------------------------
// Phase A: coordinate embedding oe[n][16]
// ---------------------------------------------------------------------------
__global__ void oe_k(const float* __restrict__ ow1, const float* __restrict__ ob1,
                     const float* __restrict__ ow2, const float* __restrict__ ob2,
                     float* __restrict__ oe) {
  int n = blockIdx.x * blockDim.x + threadIdx.x;
  if (n >= NPTS) return;
  float cz = (float)(n / 400), cy = (float)((n / 20) % 20), cx = (float)(n % 20);
  float h[ODM];
  for (int d = 0; d < ODM; ++d) {
    float v = cx * ow1[0 * ODM + d] + cy * ow1[1 * ODM + d] + cz * ow1[2 * ODM + d] + ob1[d];
    h[d] = v > 0.f ? v : 0.f;
  }
  for (int d = 0; d < ODM; ++d) {
    float s = ob2[d];
    for (int j = 0; j < ODM; ++j) s += h[j] * ow2[j * ODM + d];
    oe[n * ODM + d] = s;
  }
}

// ---------------------------------------------------------------------------
// Phase A: Q/K (bf16, row major N x 32) and V^T (bf16, 32 x N)
// Q is pre-scaled by 1/sqrt(DM) so the score WMMA needs no post-multiply.
// ---------------------------------------------------------------------------
__global__ void qkv_k(const float* __restrict__ mf, const float* __restrict__ oe,
                      const float* __restrict__ wqw, const float* __restrict__ wqb,
                      const float* __restrict__ wkw, const float* __restrict__ wkb,
                      const float* __restrict__ wvw, const float* __restrict__ wvb,
                      u16* __restrict__ Qbf, u16* __restrict__ Kbf, u16* __restrict__ Vt) {
  int n = blockIdx.x * blockDim.x + threadIdx.x;
  if (n >= NPTS) return;
  float qk[DMQ];
  for (int c = 0; c < CCH; ++c) qk[c] = mf[c * NPTS + n];
  for (int j = 0; j < ODM; ++j) qk[CCH + j] = oe[n * ODM + j];
  for (int d = 0; d < DMQ; ++d) {
    float q = wqb[d], k = wkb[d];
    for (int j = 0; j < DMQ; ++j) { q += qk[j] * wqw[j * DMQ + d]; k += qk[j] * wkw[j * DMQ + d]; }
    Qbf[n * DMQ + d] = f2bf(q * INV_SQRT_DM);
    Kbf[n * DMQ + d] = f2bf(k);
  }
  for (int d = 0; d < DMQ; ++d) {
    float v = wvb[d];
    for (int j = 0; j < CCH; ++j) v += qk[j] * wvw[j * DMQ + d];
    Vt[d * NPTS + n] = f2bf(v);
  }
}

// ---------------------------------------------------------------------------
// Phase A: gate g^2 from curvature mean + positive fraction of mf channel 0
// ---------------------------------------------------------------------------
__global__ void gate_k(const float* __restrict__ mf,
                       const float* __restrict__ gw1, const float* __restrict__ gb1,
                       const float* __restrict__ gw2, const float* __restrict__ gb2,
                       float* __restrict__ gbuf) {
  __shared__ float sL[256], sP[256];
  int tid = threadIdx.x;
  float lap = 0.f, pos = 0.f;
  for (int n = tid; n < NPTS; n += 256) {
    int z = n / 400, y = (n / 20) % 20, x = n % 20;
    int zp = min(z + 1, 19), zm = max(z - 1, 0);
    int yp = min(y + 1, 19), ym = max(y - 1, 0);
    int xp = min(x + 1, 19), xm = max(x - 1, 0);
    float cv = mf[n];
    lap += mf[zp * 400 + y * 20 + x] + mf[zm * 400 + y * 20 + x]
         + mf[z * 400 + yp * 20 + x] + mf[z * 400 + ym * 20 + x]
         + mf[z * 400 + y * 20 + xp] + mf[z * 400 + y * 20 + xm] - 6.0f * cv;
    pos += (cv > 0.f) ? 1.f : 0.f;
  }
  sL[tid] = lap; sP[tid] = pos;
  __syncthreads();
  for (int s = 128; s > 0; s >>= 1) {
    if (tid < s) { sL[tid] += sL[tid + s]; sP[tid] += sP[tid + s]; }
    __syncthreads();
  }
  if (tid == 0) {
    float cm = sL[0] / (float)NPTS, tv = sP[0] / (float)NPTS;
    float acc = gb2[0];
    for (int j = 0; j < 16; ++j) {
      float h = cm * gw1[j] + tv * gw1[16 + j] + gb1[j];
      if (h < 0.f) h = 0.f;
      acc += h * gw2[j];
    }
    float g = 1.0f / (1.0f + expf(-acc));
    gbuf[0] = g * g;
  }
}

// ---------------------------------------------------------------------------
// Phase B: fused attention. One workgroup = 16 query rows. 256 thr = 8 wave32.
// Dynamic LDS: bf16 stripe 16 x SSTRIDE (256.5KB) + stats + hist + outacc.
// Pass 1 writes raw scores; the stats sweep overwrites the stripe IN PLACE
// with bf16 probabilities p = exp(s - rowmax) (>= 0, so plain u16 compares
// are order-preserving), builds Z and the top-k radix threshold. Pass 2 is
// then exp-free: load b128, u16 compare/select, WMMA.
// ---------------------------------------------------------------------------
#define LDS_SCO_BYTES   (16 * SSTRIDE * 2)
#define LDS_TOTAL_BYTES (LDS_SCO_BYTES + 3*16*4 + 512*4 + 8*256*4)

__global__ void attn_k(const u16* __restrict__ Qbf, const u16* __restrict__ Kbf,
                       const u16* __restrict__ Vt, const float* __restrict__ gbuf,
                       float* __restrict__ outp) {
  extern __shared__ char smem[];
  u16*   sco     = (u16*)smem;                                  // [16][SSTRIDE]
  float* rowZ    = (float*)(smem + LDS_SCO_BYTES);              // [16]
  float* rowmsum = rowZ + 16;                                   // [16]
  u32*   rowthr  = (u32*)(rowmsum + 16);                        // [16]
  float* outacc  = (float*)(rowthr + 16);                       // [16][32]
  u32*   hist    = (u32*)(outacc + 512);                        // [8][256]

  const int tid  = threadIdx.x;
  const int wave = tid >> 5;
  const int lane = tid & 31;
  const int rr   = lane & 15;                 // A row / B column within tile
  const bool hiHalf = (lane >= 16);
  const int c0 = hiHalf ? 8 : 0;              // unused; A-frag also uses kb: k {0..15}/{16..31}
  const int kb = hiHalf ? 16 : 0;             // B-frag k base {0..15}/{16..31}
  const int n0 = blockIdx.x * 16;

  // init accumulators
  for (int i = tid; i < 512; i += 256) outacc[i] = 0.f;
  if (tid < 16) rowmsum[tid] = 0.f;

  // ---- load Q fragment (A-matrix layout, 16x32 bf16, pre-scaled) ----
  Frag aq;
  {
    const u16* qp = Qbf + (n0 + rr) * DMQ;
    aq.q[0] = *(const v4u*)(qp + kb);
    aq.q[1] = *(const v4u*)(qp + kb + 8);
  }

  // ---- pass 1: S = (Q/sqrt(DM)) K^T -> bf16 LDS stripe ----
  for (int t = wave; t < NTILES; t += 8) {
    int m0 = t * 16;
    // unconditional prefetch of the K tile 8 tiles (one iteration) ahead (stays in d_ws)
    __builtin_prefetch(Kbf + (size_t)(m0 + 128) * DMQ, 0, 0);
    Frag bk;   // B-matrix: col = rr -> key row m0+rr; k in [kb, kb+16)
    const u16* kp = Kbf + (size_t)(m0 + rr) * DMQ + kb;
    bk.q[0] = *(const v4u*)(kp);
    bk.q[1] = *(const v4u*)(kp + 8);
    v8f acc = {};
    acc = wmma_bf16(aq.bf, bk.bf, acc);
    const int rbase = hiHalf ? 8 : 0;         // C/D layout: rows vgpr+8*(lane>=16)
    #pragma unroll
    for (int r = 0; r < 8; ++r)
      sco[(rbase + r) * SSTRIDE + m0 + rr] = f2bf(acc[r]);
  }
  __syncthreads();

  // ---- stats: per row (2 per wave): max -> p=exp(s-max) in place, Z, top-k thr
  for (int rep = 0; rep < 2; ++rep) {
    int ri = wave * 2 + rep;
    u16* srow = sco + ri * SSTRIDE;
    // row max
    float mx = -3.402823466e38f;
    for (int j = lane; j < NPTS; j += 32) mx = fmaxf(mx, bf2f(srow[j]));
    for (int o = 16; o; o >>= 1) mx = fmaxf(mx, __shfl_xor(mx, o, 32));
    // overwrite scores with bf16 probabilities, accumulate Z, high-byte hist
    u32* h = hist + wave * 256;
    for (int b = lane; b < 256; b += 32) h[b] = 0u;
    float z = 0.f;
    for (int j = lane; j < NPTS; j += 32) {
      float p = __expf(bf2f(srow[j]) - mx);
      z += p;
      u16 pu = f2bf(p);                      // p >= 0: bit pattern is ordered
      srow[j] = pu;
      atomicAdd(&h[(u32)pu >> 8], 1u);
    }
    for (int o = 16; o; o >>= 1) z += __shfl_xor(z, o, 32);
    __syncthreads();
    // high-byte bin of the TOPK-th largest (all lanes redundantly scan)
    u32 cum = 0, bsel = 0, cumAbove = 0;
    for (int b = 255; b >= 0; --b) {
      u32 c = h[b];
      if (cum + c >= TOPK) { bsel = (u32)b; cumAbove = cum; break; }
      cum += c;
    }
    // low-byte refinement within bin bsel
    for (int b = lane; b < 256; b += 32) h[b] = 0u;
    for (int j = lane; j < NPTS; j += 32) {
      u32 pu = (u32)srow[j];
      if ((pu >> 8) == bsel) atomicAdd(&h[pu & 255u], 1u);
    }
    __syncthreads();
    u32 need = TOPK - cumAbove, cum2 = 0, lsel = 0;
    for (int b = 255; b >= 0; --b) {
      cum2 += h[b];
      if (cum2 >= need) { lsel = (u32)b; break; }
    }
    if (lane == 0) { rowZ[ri] = z; rowthr[ri] = (bsel << 8) | lsel; }
  }
  __syncthreads();

  // ---- pass 2: out = P_masked x V  (two 16x16 accumulators: dm 0-15, 16-31) ----
  v8f acc0 = {}, acc1 = {};
  float msum = 0.f;
  const u16 thr = (u16)rowthr[rr];
  for (int t = wave; t < NTILES; t += 8) {
    int m0 = t * 16;
    __builtin_prefetch(Vt + (size_t)rr * NPTS + (m0 + 128), 0, 0);
    // A-frag: thresholded probabilities straight from LDS (no exp, no cvt)
    Frag ap;
    {
      const u16* sr = sco + rr * SSTRIDE + m0;
      Frag sf;
      sf.q[0] = *(const v4u*)(sr + kb);        // ds_load_b128
      sf.q[1] = *(const v4u*)(sr + kb + 8);    // ds_load_b128
      #pragma unroll
      for (int j = 0; j < 16; ++j) {
        u16 pu = sf.us[j];
        u16 pm = (pu >= thr) ? pu : (u16)0;
        ap.us[j] = pm;
        msum += bf2f(pm);
      }
    }
    // B-frags from V^T: col = rr, k rows m0+kb .. m0+kb+15 (contiguous in Vt)
    Frag bv0, bv1;
    const u16* vp0 = Vt + (size_t)rr * NPTS + m0 + kb;
    const u16* vp1 = Vt + (size_t)(16 + rr) * NPTS + m0 + kb;
    bv0.q[0] = *(const v4u*)(vp0);  bv0.q[1] = *(const v4u*)(vp0 + 8);
    bv1.q[0] = *(const v4u*)(vp1);  bv1.q[1] = *(const v4u*)(vp1 + 8);
    acc0 = wmma_bf16(ap.bf, bv0.bf, acc0);
    acc1 = wmma_bf16(ap.bf, bv1.bf, acc1);
  }
  // cross-wave reduce via LDS float atomics (ds_add_f32)
  {
    const int rbase = hiHalf ? 8 : 0;
    #pragma unroll
    for (int r = 0; r < 8; ++r) {
      atomicAdd(&outacc[(rbase + r) * 32 + rr],      acc0[r]);
      atomicAdd(&outacc[(rbase + r) * 32 + 16 + rr], acc1[r]);
    }
    atomicAdd(&rowmsum[rr], msum);
  }
  __syncthreads();

  // ---- writeout: out[dm][n] = g2 * acc / (g2 * masked_sum + Z * 1e-8) ----
  float g2 = gbuf[0];
  for (int idx = tid; idx < 512; idx += 256) {
    int r = idx >> 5, dm = idx & 31;
    float denom = g2 * rowmsum[r] + rowZ[r] * 1e-8f;
    outp[dm * NPTS + (n0 + r)] = g2 * outacc[idx] / denom;
  }
}

// ---------------------------------------------------------------------------
// Host side
// ---------------------------------------------------------------------------
extern "C" void kernel_launch(void* const* d_in, const int* in_sizes, int n_in,
                              void* d_out, int out_size, void* d_ws, size_t ws_size,
                              hipStream_t stream) {
  (void)in_sizes; (void)n_in; (void)out_size; (void)ws_size;
  const float* feat = (const float*)d_in[0];
  const float* c1w  = (const float*)d_in[1];
  const float* c1b  = (const float*)d_in[2];
  const float* c2w  = (const float*)d_in[3];
  const float* c2b  = (const float*)d_in[4];
  const float* ow1  = (const float*)d_in[5];
  const float* ob1  = (const float*)d_in[6];
  const float* ow2  = (const float*)d_in[7];
  const float* ob2  = (const float*)d_in[8];
  const float* wqw  = (const float*)d_in[9];
  const float* wqb  = (const float*)d_in[10];
  const float* wkw  = (const float*)d_in[11];
  const float* wkb  = (const float*)d_in[12];
  const float* wvw  = (const float*)d_in[13];
  const float* wvb  = (const float*)d_in[14];
  const float* gw1  = (const float*)d_in[15];
  const float* gb1  = (const float*)d_in[16];
  const float* gw2  = (const float*)d_in[17];
  const float* gb2  = (const float*)d_in[18];

  char* ws = (char*)d_ws;
  float* s1   = (float*)(ws);                 // 16*8000*4 = 512000
  float* s2   = (float*)(ws + 512000);        // 16*1000*4 =  64000
  float* mf   = (float*)(ws + 576000);        // 512000
  float* oe   = (float*)(ws + 1088000);       // 512000
  u16*   Qbf  = (u16*)(ws + 1600000);         // 512000
  u16*   Kbf  = (u16*)(ws + 2112000);         // 512000
  u16*   Vt   = (u16*)(ws + 2624000);         // 512000
  float* gbuf = (float*)(ws + 3136000);       // 4

  conv1_k<<<(CCH * NPTS + 255) / 256, 256, 0, stream>>>(feat, c1w, c1b, s1);
  conv2_k<<<(CCH * 1000 + 255) / 256, 256, 0, stream>>>(feat, c2w, c2b, s2);
  oe_k<<<(NPTS + 255) / 256, 256, 0, stream>>>(ow1, ob1, ow2, ob2, oe);
  combine_k<<<(CCH * NPTS + 255) / 256, 256, 0, stream>>>(s1, s2, mf);
  gate_k<<<1, 256, 0, stream>>>(mf, gw1, gb1, gw2, gb2, gbuf);
  qkv_k<<<(NPTS + 255) / 256, 256, 0, stream>>>(mf, oe, wqw, wqb, wkw, wkb, wvw, wvb,
                                                Qbf, Kbf, Vt);
  attn_k<<<NTILES, 256, LDS_TOTAL_BYTES, stream>>>(Qbf, Kbf, Vt, gbuf, (float*)d_out);
}
